// EvidentialGSL_31825707664070
// MI455X (gfx1250) — matrix operand order repaired and checked by reading, not compiled, not run on hardware
//
#include <hip/hip_runtime.h>
#include <math.h>

#define N_ROWS 8192
#define D_DIM  768
#define H1_DIM 512
#define H2_DIM 256
#define TOPK   5

typedef __attribute__((ext_vector_type(16))) _Float16 v16h;
typedef __attribute__((ext_vector_type(8)))  _Float16 v8h;
typedef __attribute__((ext_vector_type(8)))  float    v8f;

__device__ __forceinline__ float gelu_f(float x) {
  return 0.5f * x * (1.0f + erff(x * 0.70710678118654752f));
}
__device__ __forceinline__ float softplus_f(float x) {
  if (x > 20.f)  return x;
  if (x < -20.f) return expf(x);
  return log1pf(expf(x));
}
__device__ __forceinline__ float sigmoid_f(float x) { return 1.f / (1.f + expf(-x)); }

// ---- WMMA fragment loaders (gfx1250 wave32, 16-bit operands in contiguous k = 16*g + e order) ----
// A: 16x32 (MxK). lane m = row; lane group g: K = g*16 + e
__device__ __forceinline__ v16h frag_a(const _Float16* __restrict__ base, int ld,
                                       int row0, int k0, int m, int g) {
  const _Float16* p = base + (size_t)(row0 + m) * (size_t)ld + (size_t)(k0 + g * 16);
  v8h lo = *(const v8h*)p;
  v8h hi = *(const v8h*)(p + 8);
  return __builtin_shufflevector(lo, hi, 0,1,2,3,4,5,6,7,8,9,10,11,12,13,14,15);
}
// B: 32x16 (KxN), supplied as BT[n][k] row-major. lane m = column; group g: K = g*16 + e
__device__ __forceinline__ v16h frag_b(const _Float16* __restrict__ bt, int ld,
                                       int col0, int k0, int m, int g) {
  return *(const v16h*)(bt + (size_t)(col0 + m) * (size_t)ld + (size_t)(k0 + g * 16));
}
#define WMMA16(a, b, c) \
  __builtin_amdgcn_wmma_f32_16x16x32_f16(false, (a), false, (b), (short)0, (c), false, false)

__device__ __forceinline__ v8f v8f_zero() {
  v8f z = {0.f, 0.f, 0.f, 0.f, 0.f, 0.f, 0.f, 0.f};
  return z;
}

// ---------------- generic f16 WMMA GEMM: C_f16 = epi(A @ B + bias) [+resid] ----------------
// block 256 = 8 waves (4 M x 2 N); wave tile 32x64 (2x4 WMMA); workgroup tile 128x128
__global__ __launch_bounds__(256) void gemm_f16_kernel(
    const _Float16* __restrict__ Amat, int lda,
    const _Float16* __restrict__ BT, int ldbt,
    _Float16* __restrict__ C, int ldc,
    const float* __restrict__ bias,
    const float* __restrict__ resid, int ldres,
    int K, int act) {
  const int lane = threadIdx.x & 31;
  const int wave = threadIdx.x >> 5;
  const int wm = wave >> 1, wn = wave & 1;
  const int m = lane & 15, g = lane >> 4;
  const int rowBase = blockIdx.x * 128 + wm * 32;
  const int colBase = blockIdx.y * 128 + wn * 64;
  v8f acc[2][4];
  for (int i = 0; i < 2; i++)
    for (int j = 0; j < 4; j++) acc[i][j] = v8f_zero();
  for (int k0 = 0; k0 < K; k0 += 32) {
    v16h a0 = frag_a(Amat, lda, rowBase, k0, m, g);
    v16h a1 = frag_a(Amat, lda, rowBase + 16, k0, m, g);
    v16h b0 = frag_b(BT, ldbt, colBase, k0, m, g);
    v16h b1 = frag_b(BT, ldbt, colBase + 16, k0, m, g);
    v16h b2 = frag_b(BT, ldbt, colBase + 32, k0, m, g);
    v16h b3 = frag_b(BT, ldbt, colBase + 48, k0, m, g);
    acc[0][0] = WMMA16(a0, b0, acc[0][0]);
    acc[0][1] = WMMA16(a0, b1, acc[0][1]);
    acc[0][2] = WMMA16(a0, b2, acc[0][2]);
    acc[0][3] = WMMA16(a0, b3, acc[0][3]);
    acc[1][0] = WMMA16(a1, b0, acc[1][0]);
    acc[1][1] = WMMA16(a1, b1, acc[1][1]);
    acc[1][2] = WMMA16(a1, b2, acc[1][2]);
    acc[1][3] = WMMA16(a1, b3, acc[1][3]);
  }
  for (int mi = 0; mi < 2; mi++) {
    for (int ni = 0; ni < 4; ni++) {
      union { v8f v; float f[8]; } u;
      u.v = acc[mi][ni];
      for (int j = 0; j < 8; j++) {
        int row = rowBase + mi * 16 + g * 8 + j;
        int col = colBase + ni * 16 + m;
        float v = u.f[j];
        if (bias) v += bias[col];
        if (act)  v = gelu_f(v);
        if (resid) v += resid[(size_t)row * ldres + col];
        C[(size_t)row * ldc + col] = (_Float16)v;
      }
    }
  }
}

// ---------------- fused S-GEMM + blend + row top-5 ----------------
// block 128 = 4 waves; 16 rows per block; sweep columns in 256-wide chunks (64 cols/wave).
__global__ __launch_bounds__(128) void topk_kernel(
    const _Float16* __restrict__ XWh, const _Float16* __restrict__ Xh,
    const float* __restrict__ A, const float* __restrict__ ra,
    int* __restrict__ tidx, float* __restrict__ tval) {
  __shared__ float sbuf[16 * 256];
  const int lane = threadIdx.x & 31;
  const int wave = threadIdx.x >> 5;
  const int m = lane & 15, g = lane >> 4;
  const int row0 = blockIdx.x * 16;
  const float al = sigmoid_f(ra[0]);
  float bv[TOPK];
  int bi[TOPK];
  for (int t = 0; t < TOPK; t++) { bv[t] = -1e30f; bi[t] = 0; }

  for (int cb = 0; cb < N_ROWS; cb += 256) {
    // prefetch the 16x256 block of A consumed by this chunk's epilogue
    {
      int rl = threadIdx.x >> 3;   // 0..15
      int seg = threadIdx.x & 7;   // 0..7 (8 x 32 floats = 256 cols)
      __builtin_prefetch(&A[(size_t)(row0 + rl) * N_ROWS + cb + seg * 32], 0, 1);
    }
    const int colBase = cb + wave * 64;
    v8f c[4];
    c[0] = v8f_zero(); c[1] = v8f_zero(); c[2] = v8f_zero(); c[3] = v8f_zero();
    for (int k0 = 0; k0 < D_DIM; k0 += 32) {
      v16h a  = frag_a(XWh, D_DIM, row0, k0, m, g);
      v16h b0 = frag_b(Xh, D_DIM, colBase, k0, m, g);
      v16h b1 = frag_b(Xh, D_DIM, colBase + 16, k0, m, g);
      v16h b2 = frag_b(Xh, D_DIM, colBase + 32, k0, m, g);
      v16h b3 = frag_b(Xh, D_DIM, colBase + 48, k0, m, g);
      c[0] = WMMA16(a, b0, c[0]);
      c[1] = WMMA16(a, b1, c[1]);
      c[2] = WMMA16(a, b2, c[2]);
      c[3] = WMMA16(a, b3, c[3]);
    }
    for (int ni = 0; ni < 4; ni++) {
      union { v8f v; float f[8]; } u;
      u.v = c[ni];
      for (int j = 0; j < 8; j++) {
        int rl = g * 8 + j;
        int col = colBase + ni * 16 + m;
        float Aij = __builtin_nontemporal_load(&A[(size_t)(row0 + rl) * N_ROWS + col]);
        sbuf[rl * 256 + (col - cb)] = al * Aij + (1.f - al) * fmaxf(u.f[j], 0.f);
      }
    }
    __syncthreads();
    if (threadIdx.x < 16) {
      const int rl = threadIdx.x;
      for (int q = 0; q < 256; q++) {
        float v = sbuf[rl * 256 + q];
        if (v > bv[TOPK - 1]) {
          int jdx = cb + q;
          int p = TOPK - 1;
          while (p > 0 && v > bv[p - 1]) { bv[p] = bv[p - 1]; bi[p] = bi[p - 1]; p--; }
          bv[p] = v; bi[p] = jdx;
        }
      }
    }
    __syncthreads();
  }
  if (threadIdx.x < 16) {
    int r = row0 + threadIdx.x;
    for (int t = 0; t < TOPK; t++) { tidx[r * TOPK + t] = bi[t]; tval[r * TOPK + t] = bv[t]; }
  }
}

// ---------------- sparse symmetrized aggregation: deg & Acc = Ag @ (X*G) ----------------
__global__ __launch_bounds__(128) void sparse_kernel(
    const int* __restrict__ tidx, const float* __restrict__ tval,
    const _Float16* __restrict__ XWh, const _Float16* __restrict__ Xh,
    const _Float16* __restrict__ XGh,
    const float* __restrict__ A, const float* __restrict__ ra,
    float* __restrict__ Acc, float* __restrict__ deg) {
  const int lane = threadIdx.x & 31;
  const int wid = (blockIdx.x * blockDim.x + threadIdx.x) >> 5;
  if (wid >= N_ROWS * TOPK) return;
  const int r = wid / TOPK;
  const int c = tidx[wid];
  const float val = tval[wid];
  if (c == r) return;  // diagonal removed by (1 - eye)
  const float al = sigmoid_f(ra[0]);
  // forward edge: Ag[r][c] = Ab[r][c]
  if (lane == 0) atomicAdd(&deg[r], val);
  for (int d = lane; d < D_DIM; d += 32)
    atomicAdd(&Acc[(size_t)r * D_DIM + d], val * (float)XGh[(size_t)c * D_DIM + d]);
  // transpose edge Ag[c][r] = Ab[c][r], unless r is already in top5(c) (dedup)
  int chk = (lane < TOPK) ? tidx[c * TOPK + lane] : -1;
  if (__ballot(chk == r) == 0ULL) {
    float dotp = 0.f;
    const _Float16* xwc = XWh + (size_t)c * D_DIM;
    const _Float16* xr  = Xh + (size_t)r * D_DIM;
    for (int k = lane; k < D_DIM; k += 32) dotp += (float)xwc[k] * (float)xr[k];
    for (int o = 16; o > 0; o >>= 1) dotp += __shfl_down(dotp, o, 32);
    float abt = al * A[(size_t)c * N_ROWS + r] + (1.f - al) * fmaxf(dotp, 0.f);
    abt = __shfl(abt, 0, 32);
    if (lane == 0) atomicAdd(&deg[c], abt);
    for (int d = lane; d < D_DIM; d += 32)
      atomicAdd(&Acc[(size_t)c * D_DIM + d], abt * (float)XGh[(size_t)r * D_DIM + d]);
  }
}

// ---------------- NIG head tail: r = H2 @ wh + bh; evidential transforms ----------------
__global__ void head_kernel(const _Float16* __restrict__ H2h, const float* __restrict__ wh,
                            const float* __restrict__ bh, const float* __restrict__ gam,
                            float* __restrict__ out, float* __restrict__ G, int head) {
  int i = blockIdx.x * blockDim.x + threadIdx.x;
  if (i >= N_ROWS) return;
  float a0 = bh[0], a1 = bh[1], a2 = bh[2], a3 = bh[3];
  const _Float16* h = H2h + (size_t)i * H2_DIM;
  for (int k = 0; k < H2_DIM; k++) {
    float hv = (float)h[k];
    a0 += hv * wh[k * 4 + 0];
    a1 += hv * wh[k * 4 + 1];
    a2 += hv * wh[k * 4 + 2];
    a3 += hv * wh[k * 4 + 3];
  }
  float mv = a0;
  float vv = softplus_f(a1) + 1e-6f;
  float av = softplus_f(a2) + 1.0f + 1e-6f;
  float bvv = softplus_f(a3) + 1e-6f;
  size_t off = (size_t)head * 4 * N_ROWS;
  out[off + 0 * N_ROWS + i] = mv;
  out[off + 1 * N_ROWS + i] = vv;
  out[off + 2 * N_ROWS + i] = av;
  out[off + 3 * N_ROWS + i] = bvv;
  if (head == 0) {
    float u0 = bvv / fmaxf(av - 1.0f, 1e-8f);
    G[i] = 1.0f - gam[0] * sigmoid_f(u0);
  }
}

// ---------------- small element-wise kernels ----------------
__global__ void cvt_f16_kernel(const float* __restrict__ in, _Float16* __restrict__ o, long long n) {
  long long i = blockIdx.x * (long long)blockDim.x + threadIdx.x;
  if (i < n) o[i] = (_Float16)in[i];
}
__global__ void cvtT_f16_kernel(const float* __restrict__ in, _Float16* __restrict__ o, int K, int Nm) {
  long long i = blockIdx.x * (long long)blockDim.x + threadIdx.x;
  long long tot = (long long)K * Nm;
  if (i < tot) {
    int k = (int)(i / Nm), n = (int)(i % Nm);
    o[(size_t)n * K + k] = (_Float16)in[i];
  }
}
__global__ void xg_kernel(const float* __restrict__ X, const float* __restrict__ G,
                          _Float16* __restrict__ XGh) {
  long long i = blockIdx.x * (long long)blockDim.x + threadIdx.x;
  if (i < (long long)N_ROWS * D_DIM) {
    int r = (int)(i / D_DIM);
    XGh[i] = (_Float16)(X[i] * G[r]);
  }
}
__global__ void scale_kernel(const float* __restrict__ Acc, const float* __restrict__ deg,
                             _Float16* __restrict__ Th) {
  long long i = blockIdx.x * (long long)blockDim.x + threadIdx.x;
  if (i < (long long)N_ROWS * D_DIM) {
    int r = (int)(i / D_DIM);
    Th[i] = (_Float16)(Acc[i] / fmaxf(deg[r], 1e-8f));
  }
}

extern "C" void kernel_launch(void* const* d_in, const int* in_sizes, int n_in,
                              void* d_out, int out_size, void* d_ws, size_t ws_size,
                              hipStream_t stream) {
  (void)in_sizes; (void)n_in; (void)out_size; (void)ws_size;
  const float* X     = (const float*)d_in[0];
  const float* A     = (const float*)d_in[1];
  const float* W_gm  = (const float*)d_in[2];
  const float* ra    = (const float*)d_in[3];
  const float* gam   = (const float*)d_in[4];
  const float* ih_w1 = (const float*)d_in[5];
  const float* ih_b1 = (const float*)d_in[6];
  const float* ih_w2 = (const float*)d_in[7];
  const float* ih_b2 = (const float*)d_in[8];
  const float* ih_wh = (const float*)d_in[9];
  const float* ih_bh = (const float*)d_in[10];
  const float* gcn_w = (const float*)d_in[11];
  const float* gcn_b = (const float*)d_in[12];
  const float* fh_w1 = (const float*)d_in[13];
  const float* fh_b1 = (const float*)d_in[14];
  const float* fh_w2 = (const float*)d_in[15];
  const float* fh_b2 = (const float*)d_in[16];
  const float* fh_wh = (const float*)d_in[17];
  const float* fh_bh = (const float*)d_in[18];
  float* out = (float*)d_out;

  const size_t N = N_ROWS, D = D_DIM, H1 = H1_DIM, H2 = H2_DIM;
  size_t off = 0;
  auto alloc = [&](size_t bytes) -> void* {
    size_t p = (off + 255) & ~(size_t)255;
    off = p + bytes;
    return (void*)((char*)d_ws + p);
  };
  _Float16* Xh   = (_Float16*)alloc(N * D * 2);
  _Float16* XWh  = (_Float16*)alloc(N * D * 2);
  _Float16* XGh  = (_Float16*)alloc(N * D * 2);
  _Float16* H1h  = (_Float16*)alloc(N * H1 * 2);
  _Float16* H2h  = (_Float16*)alloc(N * H2 * 2);
  _Float16* WgT  = (_Float16*)alloc(D * D * 2);
  _Float16* W1aT = (_Float16*)alloc(D * H1 * 2);
  _Float16* W2aT = (_Float16*)alloc(H1 * H2 * 2);
  _Float16* WgcT = (_Float16*)alloc(D * D * 2);
  _Float16* W1bT = (_Float16*)alloc(D * H1 * 2);
  _Float16* W2bT = (_Float16*)alloc(H1 * H2 * 2);
  float* Gv   = (float*)alloc(N * 4);
  int*   tidx = (int*)alloc(N * TOPK * 4);
  float* tval = (float*)alloc(N * TOPK * 4);
  float* deg  = (float*)alloc(N * 4);
  float* Acc  = (float*)alloc(N * D * 4);

  auto cvt = [&](const float* in, _Float16* o, size_t n) {
    cvt_f16_kernel<<<dim3((unsigned)((n + 255) / 256)), 256, 0, stream>>>(in, o, (long long)n);
  };
  auto cvtT = [&](const float* in, _Float16* o, int K, int Nm) {
    long long tot = (long long)K * Nm;
    cvtT_f16_kernel<<<dim3((unsigned)((tot + 255) / 256)), 256, 0, stream>>>(in, o, K, Nm);
  };
  auto gemm = [&](const _Float16* Am, int lda, const _Float16* BT, int ldbt,
                  _Float16* C, int ldc, const float* bias, const float* resid, int ldres,
                  int M, int Nm, int K, int act) {
    dim3 grid((unsigned)(M / 128), (unsigned)(Nm / 128));
    gemm_f16_kernel<<<grid, 256, 0, stream>>>(Am, lda, BT, ldbt, C, ldc, bias, resid, ldres, K, act);
  };

  // precision conversion + weight transposes
  cvt(X, Xh, N * D);
  cvtT(W_gm, WgT, D_DIM, D_DIM);
  cvtT(ih_w1, W1aT, D_DIM, H1_DIM);
  cvtT(ih_w2, W2aT, H1_DIM, H2_DIM);
  cvtT(gcn_w, WgcT, D_DIM, D_DIM);
  cvtT(fh_w1, W1bT, D_DIM, H1_DIM);
  cvtT(fh_w2, W2bT, H1_DIM, H2_DIM);

  // XW = X @ W_gm
  gemm(Xh, D_DIM, WgT, D_DIM, XWh, D_DIM, nullptr, nullptr, 0, N_ROWS, D_DIM, D_DIM, 0);
  // NIG head 0
  gemm(Xh, D_DIM, W1aT, D_DIM, H1h, H1_DIM, ih_b1, nullptr, 0, N_ROWS, H1_DIM, D_DIM, 1);
  gemm(H1h, H1_DIM, W2aT, H1_DIM, H2h, H2_DIM, ih_b2, nullptr, 0, N_ROWS, H2_DIM, H1_DIM, 1);
  head_kernel<<<dim3(N_ROWS / 256), 256, 0, stream>>>(H2h, ih_wh, ih_bh, gam, out, Gv, 0);
  // X * G
  xg_kernel<<<dim3((unsigned)((N * D + 255) / 256)), 256, 0, stream>>>(X, Gv, XGh);
  // fused S + blend + top-5 per row
  topk_kernel<<<dim3(N_ROWS / 16), 128, 0, stream>>>(XWh, Xh, A, ra, tidx, tval);
  // sparse symmetrized aggregation
  hipMemsetAsync(deg, 0, N * 4, stream);
  hipMemsetAsync(Acc, 0, N * D * 4, stream);
  sparse_kernel<<<dim3((N_ROWS * TOPK) / 4), 128, 0, stream>>>(tidx, tval, XWh, Xh, XGh, A, ra, Acc, deg);
  // Dinv scaling (reuse XWh as T)
  _Float16* Th = XWh;
  scale_kernel<<<dim3((unsigned)((N * D + 255) / 256)), 256, 0, stream>>>(Acc, deg, Th);
  // X2 = X + gelu(T @ gcn_w + gcn_b)   (reuse XGh as X2)
  _Float16* X2h = XGh;
  gemm(Th, D_DIM, WgcT, D_DIM, X2h, D_DIM, gcn_b, X, D_DIM, N_ROWS, D_DIM, D_DIM, 1);
  // NIG head 1
  gemm(X2h, D_DIM, W1bT, D_DIM, H1h, H1_DIM, fh_b1, nullptr, 0, N_ROWS, H1_DIM, D_DIM, 1);
  gemm(H1h, H1_DIM, W2bT, H1_DIM, H2h, H2_DIM, fh_b2, nullptr, 0, N_ROWS, H2_DIM, H1_DIM, 1);
  head_kernel<<<dim3(N_ROWS / 256), 256, 0, stream>>>(H2h, fh_wh, fh_bh, gam, out, Gv, 1);
}
